// CRPNet_64433099374713
// MI455X (gfx1250) — matrix operand order repaired and checked by reading, not compiled, not run on hardware
//
#include <hip/hip_runtime.h>
#include <hip/hip_bf16.h>

// ---------------------------------------------------------------------------
// CRPNet head on gfx1250: implicit-GEMM 3x3 convs via v_wmma_f32_16x16x32_f16
// Each 128-thread block (4 waves): 64 output channels x 32 pixels.
// Each wave: one 16-co tile, two 16-pixel N tiles sharing each A fragment.
// ---------------------------------------------------------------------------

typedef __attribute__((ext_vector_type(16))) _Float16 v16h;
typedef __attribute__((ext_vector_type(8)))  float    v8f;

#define NUM_LVL 5
#define ANCH 9
#define NCLS 80
#define NKPT 17
#define CIN 256
#define NB 2
#define TOTHW 8525            // 6400+1600+400+100+25
#define NROWS 153450          // NB * ANCH * TOTHW
#define FRAG 36864            // per-co-tile packed weight elems: 9*8*32*16

// ---------------------------- f32 -> f16 copy ------------------------------
__global__ void cvt_f32_f16_kernel(const float* __restrict__ src,
                                   _Float16* __restrict__ dst, int n) {
    int i = blockIdx.x * blockDim.x + threadIdx.x;
    if (i < n) dst[i] = (_Float16)src[i];
}

// --------------------- weight pack: OIHW f32 -> A-fragment f16 -------------
// dest index = (((t*9 + tau)*8 + cc)*32 + lane)*16 + e
// A-layout (16x32 f16, MxK): lane<16 -> M=lane; lane>=16 -> K+8
//   vgpr j (=e/2): K = (j>=4?16:0) + (lane>=16?8:0) + (j&3)*2 + (e&1)
__global__ void pack_w_kernel(const float* __restrict__ w,
                              _Float16* __restrict__ dst, int Cout, int total) {
    int idx = blockIdx.x * blockDim.x + threadIdx.x;
    if (idx >= total) return;
    int e = idx & 15;
    int r = idx >> 4;
    int l = r & 31;  r >>= 5;
    int cc = r & 7;  r >>= 3;
    int tau = r % 9;
    int t = r / 9;
    int j = e >> 1;
    int k = ((j & 4) << 2) | ((l & 16) >> 1) | ((j & 3) << 1) | (e & 1);
    int co = (t << 4) | (l & 15);
    int ci = (cc << 5) | k;
    float v = 0.f;
    if (co < Cout)
        v = w[(((size_t)co * CIN + ci) * 3 + tau / 3) * 3 + (tau % 3)];
    dst[idx] = (_Float16)v;
}

// --------------------------- conv3x3 via WMMA ------------------------------
// GEMM: M=Cout (16/wave), N=32 linear pixels (2 tiles/block), K=9 taps x 8x32 ci.
// A: packed weights (global, b128 loads, shared by both N tiles).
// B: im2col staged in LDS in exact B-fragment order:
//    lane -> N=lane&15; vgpr j -> K=(j>=4?16:0)+(lane>=16?8:0)+(j&3)*2+(e&1).
// D/C: M=(lane>=16?8:0)+j, N=lane&15.
template <bool HEAD, bool RELU>
__global__ void __launch_bounds__(128)
conv3x3_wmma_kernel(const _Float16* __restrict__ in,
                    const _Float16* __restrict__ wpack,
                    const float* __restrict__ bias,
                    _Float16* __restrict__ out16,
                    float* __restrict__ outF,
                    int H, int W, int tiles, int Cout,
                    int kdim, int lvlOff) {
    __shared__ v16h ldsBv[9 * 2 * 32];                // 18 KB
    __shared__ int coordY[32];
    __shared__ int coordX[32];
    _Float16* ldsB = (_Float16*)ldsBv;

    const int HW   = H * W;
    const int n    = blockIdx.z;
    const int p0   = blockIdx.x << 5;                 // 32 pixels per block
    const int tid  = threadIdx.x;
    const int lane = tid & 31;
    const int wave = tid >> 5;
    const int tile = blockIdx.y * 4 + wave;           // wave-uniform
    const int hi   = lane & 16;

    // per-block pixel coordinate table (kills per-element divisions)
    if (tid < 32) {
        int pix = p0 + tid;
        if (pix < HW) {
            int y = pix / W;
            coordY[tid] = y;
            coordX[tid] = pix - y * W;
        } else {
            coordY[tid] = -1000000;                   // forces bounds-check fail
            coordX[tid] = 0;
        }
    }

    v8f acc0 = {}, acc1 = {};
    if (tile < tiles) {
#pragma unroll
        for (int j = 0; j < 8; ++j) {
            int co = (tile << 4) + (hi >> 1) + j;
            float b = (co < Cout) ? bias[co] : 0.f;
            acc0[j] = b;
            acc1[j] = b;
        }
    }

    // per-thread loop-invariant fragment sub-index
    const int e  = tid & 15;
    const int jj = e >> 1;
    const int kb = (jj << 1) | (e & 1);

    for (int cc = 0; cc < 8; ++cc) {
        __syncthreads();                              // prev-chunk LDS reads done
        // cooperative B fill: 9 taps x 2 halves x 32 lanes x 16 = 9216 f16
        for (int u = tid >> 4; u < 9 * 2 * 32; u += 8) {
            int tau  = u / 64;
            int rem  = u - tau * 64;
            int half = rem >> 5;
            int l    = rem & 31;
            int ci   = (cc << 5) | ((e & 8) << 1) | ((l & 16) >> 1) | (e & 7);     // B-fragment K index
            int p    = (half << 4) | (l & 15);
            int yy   = coordY[p] + tau / 3 - 1;
            int xx   = coordX[p] + (tau % 3) - 1;
            float v  = 0.f;
            if (yy >= 0 && yy < H && xx >= 0 && xx < W)
                v = (float)in[((size_t)(n * CIN + ci)) * HW + yy * W + xx];
            ldsB[u * 16 + e] = (_Float16)v;
        }
        __syncthreads();
        if (tile < tiles) {                           // wave-uniform: EXEC all-1s
#pragma unroll
            for (int tau = 0; tau < 9; ++tau) {
                v16h a  = ((const v16h*)wpack)[(((size_t)tile * 9 + tau) * 8 + cc) * 32 + lane];
                v16h b0 = ldsBv[(tau * 2 + 0) * 32 + lane];
                v16h b1 = ldsBv[(tau * 2 + 1) * 32 + lane];
                acc0 = __builtin_amdgcn_wmma_f32_16x16x32_f16(
                    false, a, false, b0, (short)0, acc0, false, false);
                acc1 = __builtin_amdgcn_wmma_f32_16x16x32_f16(
                    false, a, false, b1, (short)0, acc1, false, false);
                if (cc < 7)                           // warm next chunk's A path
                    __builtin_prefetch(
                        (const void*)&((const v16h*)wpack)[(((size_t)tile * 9 + tau) * 8 + cc + 1) * 32 + lane],
                        0, 0);
            }
        }
    }

    if (tile < tiles) {
#pragma unroll
        for (int half = 0; half < 2; ++half) {
            int pixN = p0 + (half << 4) + (lane & 15);
            if (pixN < HW) {
#pragma unroll
                for (int j = 0; j < 8; ++j) {
                    int co = (tile << 4) + (hi >> 1) + j;
                    if (co < Cout) {
                        float v = half ? acc1[j] : acc0[j];
                        if (RELU) v = v > 0.f ? v : 0.f;
                        if (!HEAD) {
                            out16[((size_t)(n * Cout + co)) * HW + pixN] = (_Float16)v;
                        } else {
                            // fused _permute_to_N_HWA_K
                            int a_ = co / kdim;
                            int kk = co - a_ * kdim;
                            size_t row = (size_t)n * (ANCH * TOTHW)
                                       + (size_t)(lvlOff + pixN) * ANCH + a_;
                            outF[row * (size_t)kdim + kk] = v;
                        }
                    }
                }
            }
        }
    }
}

// --------------------- Box2KptTransform.apply_deltas -----------------------
__global__ void kpt_transform_kernel(const float* __restrict__ boxes,
                                     const float* __restrict__ stage,
                                     float* __restrict__ outk, int nrows) {
    int i = blockIdx.x * blockDim.x + threadIdx.x;    // over nrows*NKPT
    if (i >= nrows * NKPT) return;
    int r = i / NKPT;
    int k = i - r * NKPT;
    float x1 = boxes[(size_t)r * 4 + 0];
    float y1 = boxes[(size_t)r * 4 + 1];
    float x2 = boxes[(size_t)r * 4 + 2];
    float y2 = boxes[(size_t)r * 4 + 3];
    float wdt = x2 - x1, hgt = y2 - y1;
    float cx = x1 + 0.5f * wdt, cy = y1 + 0.5f * hgt;
    float dx = stage[(size_t)r * (2 * NKPT) + 2 * k];
    float dy = stage[(size_t)r * (2 * NKPT) + 2 * k + 1];
    outk[(size_t)i * 3 + 0] = dx * wdt + cx;
    outk[(size_t)i * 3 + 1] = dy * hgt + cy;
    outk[(size_t)i * 3 + 2] = 1.0f;
}

// ---------------------------------------------------------------------------
extern "C" void kernel_launch(void* const* d_in, const int* in_sizes, int n_in,
                              void* d_out, int out_size, void* d_ws, size_t ws_size,
                              hipStream_t stream) {
    (void)in_sizes; (void)n_in; (void)out_size; (void)ws_size;

    const float* feat[NUM_LVL];
    for (int i = 0; i < NUM_LVL; ++i) feat[i] = (const float*)d_in[i];
    const float* cls_w   = (const float*)d_in[5];
    const float* cls_b   = (const float*)d_in[6];
    const float* bbox_w  = (const float*)d_in[7];
    const float* bbox_b  = (const float*)d_in[8];
    const float* kpt_w   = (const float*)d_in[9];
    const float* kpt_b   = (const float*)d_in[10];
    const float* clsS_w  = (const float*)d_in[11];
    const float* clsS_b  = (const float*)d_in[12];
    const float* bbS_w   = (const float*)d_in[13];
    const float* bbS_b   = (const float*)d_in[14];
    const float* kpS_w   = (const float*)d_in[15];
    const float* kpS_b   = (const float*)d_in[16];
    const float* boxes   = (const float*)d_in[17];

    static const int LH[NUM_LVL]     = {80, 40, 20, 10, 5};
    static const int lvlOff[NUM_LVL] = {0, 6400, 8000, 8400, 8500};

    // ---- workspace carve-out ----
    char* ws = (char*)d_ws;
    size_t off = 0;
    auto take = [&](size_t bytes) -> char* {
        char* p = ws + off;
        off = (off + bytes + 255) & ~(size_t)255;
        return p;
    };
    const size_t LAYER = (size_t)16 * FRAG;           // 256-out layer: 589824 f16
    _Float16* pw_cls  = (_Float16*)take(4 * LAYER * 2);
    _Float16* pw_bbox = (_Float16*)take(4 * LAYER * 2);
    _Float16* pw_kpt  = (_Float16*)take(4 * LAYER * 2);
    _Float16* pw_clsS = (_Float16*)take((size_t)45 * FRAG * 2);   // Cout 720
    _Float16* pw_bbS  = (_Float16*)take((size_t)3  * FRAG * 2);   // Cout 36 (pad 48)
    _Float16* pw_kpS  = (_Float16*)take((size_t)20 * FRAG * 2);   // Cout 306 (pad 320)
    _Float16* feat16[NUM_LVL];
    for (int l = 0; l < NUM_LVL; ++l)
        feat16[l] = (_Float16*)take((size_t)NB * CIN * LH[l] * LH[l] * 2);
    _Float16* actA = (_Float16*)take((size_t)NB * CIN * 6400 * 2);
    _Float16* actB = (_Float16*)take((size_t)NB * CIN * 6400 * 2);
    float* kstage  = (float*)take((size_t)NROWS * (2 * NKPT) * 4);

    // ---- pack weights (f32 -> f16, A-fragment order) ----
    auto packW = [&](const float* src, _Float16* dst, int Cout, int tiles) {
        int total = tiles * FRAG;
        pack_w_kernel<<<(total + 255) / 256, 256, 0, stream>>>(src, dst, Cout, total);
    };
    const size_t RAWL = (size_t)CIN * CIN * 9;
    for (int i = 0; i < 4; ++i) {
        packW(cls_w  + i * RAWL, pw_cls  + i * LAYER, CIN, 16);
        packW(bbox_w + i * RAWL, pw_bbox + i * LAYER, CIN, 16);
        packW(kpt_w  + i * RAWL, pw_kpt  + i * LAYER, CIN, 16);
    }
    packW(clsS_w, pw_clsS, ANCH * NCLS, 45);
    packW(bbS_w,  pw_bbS,  ANCH * 4,    3);
    packW(kpS_w,  pw_kpS,  ANCH * NKPT * 2, 20);

    // ---- convert features to f16 ----
    for (int l = 0; l < NUM_LVL; ++l) {
        int n = NB * CIN * LH[l] * LH[l];
        cvt_f32_f16_kernel<<<(n + 255) / 256, 256, 0, stream>>>(feat[l], feat16[l], n);
    }

    float* out      = (float*)d_out;
    float* boxcls   = out;
    float* boxdelta = out + (size_t)NROWS * NCLS;
    float* kptout   = out + (size_t)NROWS * (NCLS + 4);

    // ---- run one tower (4 mid convs + head conv) for one level ----
    auto runChain = [&](int l, _Float16* pws, const float* biases,
                        _Float16* pwh, const float* bh, int CoutH, int tilesH,
                        int kdim, float* outF) {
        int H = LH[l], W = LH[l], HW = H * W;
        dim3 gridM((HW + 31) / 32, 4, NB);            // 16 co-tiles / 4 waves
        const _Float16* cur = feat16[l];
        _Float16* buf[2] = {actA, actB};
        for (int i = 0; i < 4; ++i) {
            conv3x3_wmma_kernel<false, true><<<gridM, 128, 0, stream>>>(
                cur, pws + (size_t)i * LAYER, biases + i * CIN,
                buf[i & 1], nullptr, H, W, 16, CIN, 0, 0);
            cur = buf[i & 1];
        }
        dim3 gridH((HW + 31) / 32, (tilesH + 3) / 4, NB);
        conv3x3_wmma_kernel<true, false><<<gridH, 128, 0, stream>>>(
            cur, pwh, bh, nullptr, outF, H, W, tilesH, CoutH, kdim, lvlOff[l]);
    };

    for (int l = 0; l < NUM_LVL; ++l) {
        runChain(l, pw_cls,  cls_b,  pw_clsS, clsS_b, ANCH * NCLS,     45, NCLS,     boxcls);
        runChain(l, pw_bbox, bbox_b, pw_bbS,  bbS_b,  ANCH * 4,         3, 4,        boxdelta);
        runChain(l, pw_kpt,  kpt_b,  pw_kpS,  kpS_b,  ANCH * NKPT * 2, 20, NKPT * 2, kstage);
    }

    int nk = NROWS * NKPT;
    kpt_transform_kernel<<<(nk + 255) / 256, 256, 0, stream>>>(boxes, kstage, kptout, NROWS);
}
